// MoELayer_9517647527955
// MI455X (gfx1250) — matrix operand order repaired and checked by reading, not compiled, not run on hardware
//
#include <hip/hip_runtime.h>
#include <hip/hip_bf16.h>
#include <math.h>

// ---------------- problem constants ----------------
#define E_  12
#define D_  1024
#define H_  2048
#define HH_ 1024
#define O_  1024
#define K_  4
#define G1_ 256
#define G2_ 128
#define B_  4096
#define R_  (B_*K_)      // total routed rows = 16384
#define SLACK_ 128       // extra rows on A-operand buffers -> unguarded tile loads

typedef __bf16 bf16;
typedef __attribute__((ext_vector_type(16))) __bf16 v16bf;
typedef __attribute__((ext_vector_type(8)))  float  v8f;
typedef __attribute__((ext_vector_type(4)))  int    v4i;

union FragAB { v16bf v; unsigned int u[8]; };

// ---- CDNA5 async LDS copy (ASYNCcnt path), with safe fallback ----
#if __has_builtin(__builtin_amdgcn_global_load_async_to_lds_b128) && \
    __has_builtin(__builtin_amdgcn_s_wait_asynccnt)
#define HAVE_ASYNC_LDS 1
#else
#define HAVE_ASYNC_LDS 0
#endif

__device__ __forceinline__ void copy16_to_lds(const bf16* g, bf16* l) {
#if HAVE_ASYNC_LDS
    __builtin_amdgcn_global_load_async_to_lds_b128(
        (__attribute__((address_space(1))) v4i*)g,
        (__attribute__((address_space(3))) v4i*)l, 0, 0);
#else
    *(uint4*)l = *(const uint4*)g;
#endif
}
__device__ __forceinline__ void wait_stage() {
#if HAVE_ASYNC_LDS
    __builtin_amdgcn_s_wait_asynccnt(0);
#endif
    __syncthreads();
}

// ---------------- transpose + f32->bf16: dst[e][n][k] = (bf16)src[e][k][n] ----------------
__global__ __launch_bounds__(256)
void transpose_f2bf(const float* __restrict__ src, bf16* __restrict__ dst, int Kdim, int Ndim) {
    __shared__ float t[32][33];
    const int e  = blockIdx.z;
    const int k0 = blockIdx.x * 32;
    const int n0 = blockIdx.y * 32;
    const float* S = src + (size_t)e * Kdim * Ndim;
    bf16*       Dd = dst + (size_t)e * Kdim * Ndim;
    const int tx = threadIdx.x & 31, ty = threadIdx.x >> 5;   // 32 x 8
    #pragma unroll
    for (int i = 0; i < 32; i += 8)
        t[ty + i][tx] = S[(size_t)(k0 + ty + i) * Ndim + n0 + tx];
    __syncthreads();
    #pragma unroll
    for (int i = 0; i < 32; i += 8)
        Dd[(size_t)(n0 + ty + i) * Kdim + k0 + tx] = (bf16)t[tx][ty + i];
}

// ---------------- fused gating network (one block per token) ----------------
__global__ __launch_bounds__(256)
void gating_kernel(const float* __restrict__ x,
                   const float* __restrict__ gw_in, const float* __restrict__ gb_in,
                   const float* __restrict__ gln1_g, const float* __restrict__ gln1_b,
                   const float* __restrict__ gw_1,  const float* __restrict__ gb_1,
                   const float* __restrict__ gln2_g,const float* __restrict__ gln2_b,
                   const float* __restrict__ gw_2,  const float* __restrict__ gb_2,
                   const float* __restrict__ gw_out,const float* __restrict__ gb_out,
                   const float* __restrict__ temperature,
                   float* __restrict__ probs_out,
                   int* __restrict__ topi, float* __restrict__ topw)
{
    __shared__ float xs[D_];
    __shared__ float t1[G1_];
    __shared__ float g2s[G2_];
    __shared__ float red[256];

    const int b = blockIdx.x, tid = threadIdx.x;
    for (int i = tid; i < D_; i += 256) xs[i] = x[(size_t)b*D_ + i];
    __syncthreads();

    float acc = gb_in[tid];
    for (int d = 0; d < D_; ++d) acc = fmaf(xs[d], gw_in[(size_t)d*G1_ + tid], acc);
    const float g0v = fmaxf(acc, 0.f);

    red[tid] = g0v; __syncthreads();
    for (int s = 128; s > 0; s >>= 1) { if (tid < s) red[tid] += red[tid+s]; __syncthreads(); }
    float mean = red[0] * (1.f/G1_); __syncthreads();
    float dce = g0v - mean;
    red[tid] = dce*dce; __syncthreads();
    for (int s = 128; s > 0; s >>= 1) { if (tid < s) red[tid] += red[tid+s]; __syncthreads(); }
    float rstd = rsqrtf(red[0]*(1.f/G1_) + 1e-5f); __syncthreads();
    t1[tid] = fmaxf(dce*rstd*gln1_g[tid] + gln1_b[tid], 0.f);
    __syncthreads();

    acc = gb_1[tid];
    for (int d = 0; d < G1_; ++d) acc = fmaf(t1[d], gw_1[d*G1_ + tid], acc);
    const float g1v = acc + g0v;
    __syncthreads();

    red[tid] = g1v; __syncthreads();
    for (int s = 128; s > 0; s >>= 1) { if (tid < s) red[tid] += red[tid+s]; __syncthreads(); }
    mean = red[0] * (1.f/G1_); __syncthreads();
    dce = g1v - mean;
    red[tid] = dce*dce; __syncthreads();
    for (int s = 128; s > 0; s >>= 1) { if (tid < s) red[tid] += red[tid+s]; __syncthreads(); }
    rstd = rsqrtf(red[0]*(1.f/G1_) + 1e-5f); __syncthreads();
    t1[tid] = fmaxf(dce*rstd*gln2_g[tid] + gln2_b[tid], 0.f);
    __syncthreads();

    if (tid < G2_) {
        float a2 = gb_2[tid];
        for (int d = 0; d < G1_; ++d) a2 = fmaf(t1[d], gw_2[d*G2_ + tid], a2);
        g2s[tid] = a2;
    }
    __syncthreads();

    if (tid < E_) {
        float l = gb_out[tid];
        for (int d = 0; d < G2_; ++d) l = fmaf(g2s[d], gw_out[d*E_ + tid], l);
        red[tid] = l / temperature[0];
    }
    __syncthreads();

    if (tid == 0) {
        float lmax = red[0];
        for (int e = 1; e < E_; ++e) lmax = fmaxf(lmax, red[e]);
        float p[E_], s = 0.f;
        for (int e = 0; e < E_; ++e) { p[e] = expf(red[e] - lmax); s += p[e]; }
        const float inv = 1.f / s;
        for (int e = 0; e < E_; ++e) { p[e] *= inv; probs_out[(size_t)b*E_ + e] = p[e]; }
        int used = 0; float tp[K_]; int ti[K_]; float psum = 0.f;
        for (int k = 0; k < K_; ++k) {
            float best = -1.f; int bi = 0;
            for (int e = 0; e < E_; ++e)
                if (!((used >> e) & 1) && p[e] > best) { best = p[e]; bi = e; }
            used |= 1 << bi; ti[k] = bi; tp[k] = best; psum += best;
        }
        const float invp = 1.f / psum;
        for (int k = 0; k < K_; ++k) { topi[b*K_+k] = ti[k]; topw[b*K_+k] = tp[k]*invp; }
    }
}

// ---------------- load-balancing loss (single block) ----------------
__global__ __launch_bounds__(256)
void loss_kernel(const float* __restrict__ probs, float* __restrict__ loss_out)
{
    __shared__ float red[256];
    __shared__ float us[E_];
    const int tid = threadIdx.x;
    float acc[E_];
    for (int e = 0; e < E_; ++e) acc[e] = 0.f;
    for (int b = tid; b < B_; b += 256)
        for (int e = 0; e < E_; ++e) acc[e] += probs[(size_t)b*E_ + e];
    for (int e = 0; e < E_; ++e) {
        red[tid] = acc[e]; __syncthreads();
        for (int s = 128; s > 0; s >>= 1) { if (tid < s) red[tid] += red[tid+s]; __syncthreads(); }
        if (tid == 0) us[e] = red[0] * (1.f/B_);
        __syncthreads();
    }
    if (tid == 0) {
        const float ideal = 1.f/E_;
        float kl = 0.f, var = 0.f, ent = 0.f;
        for (int e = 0; e < E_; ++e) {
            kl  += ideal * (logf(ideal) - logf(us[e] + 1e-8f));
            var += (us[e]-ideal)*(us[e]-ideal);
            ent -= us[e] * logf(us[e] + 1e-8f);
        }
        const float ent_loss = 1.f - ent / logf((float)E_);
        loss_out[0] = (0.5f*kl + 0.3f*var + 0.2f*ent_loss) * 0.05f;
    }
}

// ---------------- routing ----------------
__global__ void route_init(int* counts, int* cursor) {
    int t = threadIdx.x;
    if (t < E_) { counts[t] = 0; cursor[t] = 0; }
}
__global__ void route_count(const int* __restrict__ topi, int* __restrict__ counts) {
    int p = blockIdx.x * blockDim.x + threadIdx.x;
    if (p < R_) atomicAdd(&counts[topi[p]], 1);
}
__global__ void route_scan(const int* __restrict__ counts, int* __restrict__ startArr) {
    if (threadIdx.x == 0 && blockIdx.x == 0) {
        int s = 0;
        for (int e = 0; e < E_; ++e) { startArr[e] = s; s += counts[e]; }
        startArr[E_] = s;
    }
}
__global__ void route_fill(const int* __restrict__ topi, const int* __restrict__ startArr,
                           int* __restrict__ cursor, int* __restrict__ p2s) {
    int p = blockIdx.x * blockDim.x + threadIdx.x;
    if (p < R_) {
        int e = topi[p];
        int pos = atomicAdd(&cursor[e], 1);
        p2s[p] = startArr[e] + pos;
    }
}
__global__ __launch_bounds__(256)
void gather_x(const float* __restrict__ x, const int* __restrict__ p2s, bf16* __restrict__ Xg) {
    const int p = blockIdx.x;
    const int slot = p2s[p];
    const int b = p >> 2;                 // K_ == 4
    for (int i = threadIdx.x; i < D_; i += 256)
        Xg[(size_t)slot*D_ + i] = (bf16)x[(size_t)b*D_ + i];
}

// ---------------- WMMA GEMM over routed rows ----------------
// 128x128 tile per block, 8 waves (wave32) as 4(M) x 2(N); each wave 2x4 WMMA tiles.
// Double-buffered LDS; tiles staged by async DMA (ASYNCcnt) when available.
// A buffers carry SLACK_ rows so staging needs no row guard (EXEC stays full).
// Wt is pre-transposed [e][n][k] so both tiles stage as b128 copies.
// EPI: 0 = +bias ; 1 = relu(+bias) ; 2 = +bias + residual
template<int EPI, typename OutT>
__global__ __launch_bounds__(256)
void gemm_moe(const bf16* __restrict__ A, int lda,
              const bf16* __restrict__ Wt, int Kdim, int Ndim,
              const float* __restrict__ biasAll,
              const float* __restrict__ residAll,
              OutT* __restrict__ Out, int ldo,
              const int* __restrict__ startArr)
{
    __shared__ __attribute__((aligned(16))) bf16 At[2][128][40];   // [m][k]
    __shared__ __attribute__((aligned(16))) bf16 Bt[2][128][40];   // [n][k]

    const int e  = blockIdx.y >> 5;       // 32 M-tiles per expert (worst case count = B)
    const int mt = blockIdx.y & 31;
    const int rowStart = startArr[e];
    const int rowEnd   = startArr[e+1];
    const int rowBase  = rowStart + mt * 128;
    if (rowBase >= rowEnd) return;        // whole-block exit: EXEC all-ones around WMMA

    const int nb = blockIdx.x * 128;
    const bf16*  Ab   = A  + (size_t)rowBase * lda;
    const bf16*  Wb   = Wt + (size_t)e * Kdim * Ndim + (size_t)nb * Kdim;
    const float* bias = biasAll + (size_t)e * Ndim;

    const int tid  = threadIdx.x;
    const int lane = tid & 31;
    const int wave = tid >> 5;
    const int wm   = wave & 3;            // 32-row strip
    const int wn   = wave >> 2;           // 64-col strip
    const int rsel = lane & 15;
    const int half = lane >> 4;

    v8f acc[2][4];
    for (int i = 0; i < 2; ++i)
        for (int j = 0; j < 4; ++j)
            for (int q = 0; q < 8; ++q) acc[i][j][q] = 0.f;

    auto stage = [&](int buf, int kb) {
        #pragma unroll
        for (int c = tid; c < 512; c += 256) {
            const int r = c >> 2, q = c & 3;
            copy16_to_lds(Ab + (size_t)r*lda + kb + q*8, &At[buf][r][q*8]);
        }
        #pragma unroll
        for (int c = tid; c < 512; c += 256) {
            const int n = c >> 2, q = c & 3;
            copy16_to_lds(Wb + (size_t)n*Kdim + kb + q*8, &Bt[buf][n][q*8]);
        }
    };

    stage(0, 0);
    int buf = 0;
    for (int kb = 0; kb < Kdim; kb += 32) {
        wait_stage();                       // own async copies done + all waves staged
        if (kb + 32 < Kdim) stage(buf ^ 1, kb + 32);

        // A 16x32 bf16 frag: lane<16 rows, V0..3 K0..7, V4..7 K16..23; lane>=16: K+8
        FragAB afr[2], bfr[4];
        const int base0 = half * 8;
        #pragma unroll
        for (int i = 0; i < 2; ++i) {
            const int r = wm*32 + i*16 + rsel;
            #pragma unroll
            for (int p = 0; p < 8; ++p) {
                const int ks = base0 + 2*p + (p >= 4 ? 8 : 0);
                afr[i].u[p] = *(const unsigned int*)&At[buf][r][ks];
            }
        }
        // B 32x16 bf16 frag: lane<16 cols, V0..3 K0..7, V4..7 K16..23; lane>=16: K+8
        const int kOff = half * 8;
        #pragma unroll
        for (int j = 0; j < 4; ++j) {
            const int cc = wn*64 + j*16 + rsel;
            #pragma unroll
            for (int p = 0; p < 8; ++p)
                bfr[j].u[p] = *(const unsigned int*)&Bt[buf][cc][kOff + 2*p + (p >= 4 ? 8 : 0)];
        }

        #pragma unroll
        for (int i = 0; i < 2; ++i)
            #pragma unroll
            for (int j = 0; j < 4; ++j)
                acc[i][j] = __builtin_amdgcn_wmma_f32_16x16x32_bf16(
                    false, afr[i].v, false, bfr[j].v, (short)0, acc[i][j], false, false);
        buf ^= 1;
    }

    // epilogue: C/D layout lane<16 -> M=q, lane>=16 -> M=q+8; N = lane&15
    const int rhalf = half * 8;
    for (int i = 0; i < 2; ++i) {
        for (int j = 0; j < 4; ++j) {
            const int col = nb + wn*64 + j*16 + rsel;
            const float bcol = bias[col];
            #pragma unroll
            for (int q = 0; q < 8; ++q) {
                const int slot = rowBase + wm*32 + i*16 + rhalf + q;
                if (slot < rowEnd) {
                    float v = acc[i][j][q] + bcol;
                    if (EPI == 1) v = fmaxf(v, 0.f);
                    if (EPI == 2) v += residAll[(size_t)slot*ldo + col];
                    Out[(size_t)slot*ldo + col] = (OutT)v;
                }
            }
        }
    }
}

// ---------------- LayerNorm + activation (row per block, H_ = 2048) ----------------
__global__ __launch_bounds__(256)
void ln_act(const float* __restrict__ hin, bf16* __restrict__ aout,
            const float* __restrict__ gAll, const float* __restrict__ bAll,
            const int* __restrict__ startArr, int silu)
{
    __shared__ float red[256];
    const int row = blockIdx.x;
    const int tid = threadIdx.x;
    int e = 0;
    while (row >= startArr[e+1]) ++e;

    const float* hrow = hin + (size_t)row * H_;
    float lv[8], s = 0.f;
    #pragma unroll
    for (int i = 0; i < 8; ++i) { lv[i] = hrow[tid + i*256]; s += lv[i]; }
    red[tid] = s; __syncthreads();
    for (int k = 128; k > 0; k >>= 1) { if (tid < k) red[tid] += red[tid+k]; __syncthreads(); }
    const float mean = red[0] * (1.f/H_); __syncthreads();
    float vs = 0.f;
    #pragma unroll
    for (int i = 0; i < 8; ++i) { const float d = lv[i]-mean; vs += d*d; }
    red[tid] = vs; __syncthreads();
    for (int k = 128; k > 0; k >>= 1) { if (tid < k) red[tid] += red[tid+k]; __syncthreads(); }
    const float rstd = rsqrtf(red[0]*(1.f/H_) + 1e-5f);

    const float* g = gAll + (size_t)e * H_;
    const float* b = bAll + (size_t)e * H_;
    #pragma unroll
    for (int i = 0; i < 8; ++i) {
        const int h = tid + i*256;
        float v = (lv[i]-mean)*rstd*g[h] + b[h];
        v = silu ? (v / (1.f + expf(-v))) : fmaxf(v, 0.f);
        aout[(size_t)row*H_ + h] = (bf16)v;
    }
}

// ---------------- weighted combine: final[b] = sum_k w_k * y[slot(b,k)] ----------------
__global__ __launch_bounds__(256)
void combine_kernel(const float* __restrict__ yb, const int* __restrict__ p2s,
                    const float* __restrict__ topw, float* __restrict__ outF)
{
    const int b = blockIdx.x;
    const int s0 = p2s[b*K_+0], s1 = p2s[b*K_+1], s2 = p2s[b*K_+2], s3 = p2s[b*K_+3];
    const float w0 = topw[b*K_+0], w1 = topw[b*K_+1], w2 = topw[b*K_+2], w3 = topw[b*K_+3];
    for (int o = threadIdx.x; o < O_; o += 256) {
        float v = w0 * yb[(size_t)s0*O_ + o];
        v = fmaf(w1, yb[(size_t)s1*O_ + o], v);
        v = fmaf(w2, yb[(size_t)s2*O_ + o], v);
        v = fmaf(w3, yb[(size_t)s3*O_ + o], v);
        outF[(size_t)b*O_ + o] = v;
    }
}

// ---------------- launcher ----------------
extern "C" void kernel_launch(void* const* d_in, const int* in_sizes, int n_in,
                              void* d_out, int out_size, void* d_ws, size_t ws_size,
                              hipStream_t stream) {
    const float* x      = (const float*)d_in[0];
    const float* ew_in  = (const float*)d_in[1];
    const float* eb_in  = (const float*)d_in[2];
    const float* ln1_g  = (const float*)d_in[3];
    const float* ln1_b  = (const float*)d_in[4];
    const float* ew_h1  = (const float*)d_in[5];
    const float* eb_h1  = (const float*)d_in[6];
    const float* ln2_g  = (const float*)d_in[7];
    const float* ln2_b  = (const float*)d_in[8];
    const float* ew_h2  = (const float*)d_in[9];
    const float* eb_h2  = (const float*)d_in[10];
    const float* ew_out = (const float*)d_in[11];
    const float* eb_out = (const float*)d_in[12];
    const float* gw_in  = (const float*)d_in[13];
    const float* gb_in  = (const float*)d_in[14];
    const float* gln1_g = (const float*)d_in[15];
    const float* gln1_b = (const float*)d_in[16];
    const float* gw_1   = (const float*)d_in[17];
    const float* gb_1   = (const float*)d_in[18];
    const float* gln2_g = (const float*)d_in[19];
    const float* gln2_b = (const float*)d_in[20];
    const float* gw_2   = (const float*)d_in[21];
    const float* gb_2   = (const float*)d_in[22];
    const float* gw_out = (const float*)d_in[23];
    const float* gb_out = (const float*)d_in[24];
    const float* temp   = (const float*)d_in[25];

    float* outF  = (float*)d_out;
    float* lossO = outF + (size_t)B_*O_;
    float* probs = outF + (size_t)B_*O_ + 1;

    // workspace carve-up (~770 MB); A-operand buffers get SLACK_ extra rows
    char* p = (char*)d_ws;
    auto alloc = [&](size_t bytes) { void* r = (void*)p; p += (bytes + 255) & ~(size_t)255; return r; };
    bf16*  W1t = (bf16*) alloc((size_t)E_*D_ *H_ *2);   // [e][H][D]
    bf16*  W2t = (bf16*) alloc((size_t)E_*H_ *H_ *2);   // [e][H][H]
    bf16*  W3t = (bf16*) alloc((size_t)E_*H_ *HH_*2);   // [e][HH][H]
    bf16*  W4t = (bf16*) alloc((size_t)E_*HH_*O_ *2);   // [e][O][HH]
    bf16*  Xg  = (bf16*) alloc((size_t)(R_+SLACK_)*D_ *2);
    float* h0  = (float*)alloc((size_t)R_*H_ *4);
    bf16*  a1  = (bf16*) alloc((size_t)(R_+SLACK_)*H_ *2);
    float* h1  = (float*)alloc((size_t)R_*H_ *4);
    bf16*  a2  = (bf16*) alloc((size_t)(R_+SLACK_)*H_ *2);
    bf16*  h2  = (bf16*) alloc((size_t)(R_+SLACK_)*HH_*2);
    float* yb  = (float*)alloc((size_t)R_*O_ *4);
    int*   topi   = (int*)  alloc((size_t)R_*4);
    float* topw   = (float*)alloc((size_t)R_*4);
    int*   p2s    = (int*)  alloc((size_t)R_*4);
    int*   counts = (int*)  alloc(64);
    int*   startA = (int*)  alloc(64);
    int*   cursor = (int*)  alloc(64);

    // 1) weights -> bf16, transposed to [n][k] (one pass, L2-friendly tiles)
    transpose_f2bf<<<dim3(D_/32,  H_/32,  E_), 256, 0, stream>>>(ew_in,  W1t, D_,  H_ );
    transpose_f2bf<<<dim3(H_/32,  H_/32,  E_), 256, 0, stream>>>(ew_h1,  W2t, H_,  H_ );
    transpose_f2bf<<<dim3(H_/32,  HH_/32, E_), 256, 0, stream>>>(ew_h2,  W3t, H_,  HH_);
    transpose_f2bf<<<dim3(HH_/32, O_/32,  E_), 256, 0, stream>>>(ew_out, W4t, HH_, O_ );

    // 2) gating + loss
    gating_kernel<<<B_, 256, 0, stream>>>(x, gw_in, gb_in, gln1_g, gln1_b, gw_1, gb_1,
                                          gln2_g, gln2_b, gw_2, gb_2, gw_out, gb_out,
                                          temp, probs, topi, topw);
    loss_kernel<<<1, 256, 0, stream>>>(probs, lossO);

    // 3) routing + gather
    route_init <<<1, 32, 0, stream>>>(counts, cursor);
    route_count<<<(R_+255)/256, 256, 0, stream>>>(topi, counts);
    route_scan <<<1, 1, 0, stream>>>(counts, startA);
    route_fill <<<(R_+255)/256, 256, 0, stream>>>(topi, startA, cursor, p2s);
    gather_x   <<<R_, 256, 0, stream>>>(x, p2s, Xg);

    // 4) expert MLP stack (sparse over routed rows), WMMA bf16
    dim3 blk(256);
    dim3 g1(H_ /128, E_*32);
    dim3 g2(H_ /128, E_*32);
    dim3 g3(HH_/128, E_*32);
    dim3 g4(O_ /128, E_*32);
    gemm_moe<1, float><<<g1, blk, 0, stream>>>(Xg, D_, W1t, D_, H_,  eb_in,  nullptr, h0, H_,  startA);
    ln_act<<<R_, 256, 0, stream>>>(h0, a1, ln1_g, ln1_b, startA, 0);
    gemm_moe<2, float><<<g2, blk, 0, stream>>>(a1, H_, W2t, H_, H_,  eb_h1,  h0,      h1, H_,  startA);
    ln_act<<<R_, 256, 0, stream>>>(h1, a2, ln2_g, ln2_b, startA, 1);
    gemm_moe<0, bf16 ><<<g3, blk, 0, stream>>>(a2, H_, W3t, H_, HH_, eb_h2,  nullptr, h2, HH_, startA);
    gemm_moe<0, float><<<g4, blk, 0, stream>>>(h2, HH_, W4t, HH_, O_, eb_out, nullptr, yb, O_,  startA);

    // 5) combine top-k
    combine_kernel<<<B_, 256, 0, stream>>>(yb, p2s, topw, outF);

    (void)in_sizes; (void)n_in; (void)out_size; (void)ws_size;
}
